// LinearDecomMPO_42545946034399
// MI455X (gfx1250) — matrix operand order repaired and checked by reading, not compiled, not run on hardware
//
#include <hip/hip_runtime.h>

// ---------------------------------------------------------------------------
// MPO (tensor-train) linear:  out = x @ W^T + bias,  W contracted from 4 cores
//   Phase 0: split x fp32 -> bf16 hi/lo arrays (Xhi, Xlo)        (bandwidth)
//   Phase 1: merge (c0,c1)->W01[A,C,r2], (c2,c3)->W23[r2,B,D]    (tiny)
//   Phase 2: W[O,I] = sum_r2 W01 * W23, split into bf16 hi/lo    (2.1 GFLOP)
//   Phase 3: 4096^3 GEMM via v_wmma_f32_16x16x32_bf16, 3-pass
//            split precision (hi*hi + hi*lo + lo*hi), fp32 accum (137 GFLOP x3)
// All bf16 operands (128 MB) stay L2-resident (192 MB) during the GEMM;
// output is stored non-temporally so the 64 MB stream doesn't evict them.
// Wave tile 64x64: 48 WMMA per 32 b128 loads per k-step (A reused across 4
// N-subtiles, B across 4 M-subtiles) to keep the matrix pipe, not L0, limiting.
// ---------------------------------------------------------------------------

typedef __bf16 bf16_t;
typedef __attribute__((ext_vector_type(16))) __bf16 v16bf;
typedef __attribute__((ext_vector_type(8)))  __bf16 v8bf;
typedef __attribute__((ext_vector_type(8)))  float  v8f;

union Frag16 {            // lets two b128 loads land directly in fragment VGPRs
  v16bf v;
  v8bf  h[2];
};

__device__ __forceinline__ bf16_t f32_to_bf16_rne(float f) {
  unsigned u = __builtin_bit_cast(unsigned, f);
  u += 0x7FFFu + ((u >> 16) & 1u);                 // round-to-nearest-even
  unsigned short s = (unsigned short)(u >> 16);
  return __builtin_bit_cast(bf16_t, s);
}
__device__ __forceinline__ float bf16_to_f32(bf16_t b) {
  unsigned short s = __builtin_bit_cast(unsigned short, b);
  unsigned u = ((unsigned)s) << 16;
  return __builtin_bit_cast(float, u);
}

// --------------------------- Phase 0: split x -------------------------------
__global__ __launch_bounds__(256) void split_x(
    const float* __restrict__ x, bf16_t* __restrict__ Xhi,
    bf16_t* __restrict__ Xlo) {
  int idx = (blockIdx.x * 256 + threadIdx.x) * 4;  // 4 floats per thread
  float4 f = *(const float4*)(x + idx);
  float v[4] = {f.x, f.y, f.z, f.w};
  typedef __attribute__((ext_vector_type(4))) __bf16 v4bf;
  v4bf hi, lo;
#pragma unroll
  for (int j = 0; j < 4; ++j) {
    bf16_t h = f32_to_bf16_rne(v[j]);
    hi[j] = h;
    lo[j] = f32_to_bf16_rne(v[j] - bf16_to_f32(h));
  }
  *(v4bf*)(Xhi + idx) = hi;                        // re-read by GEMM: keep RT
  *(v4bf*)(Xlo + idx) = lo;
}

// --------------------------- Phase 1: core pair merges ----------------------
// c0:(1,8,8,64)  c1:(64,8,8,64)  c2:(64,8,8,64)  c3:(64,8,8,1)
__global__ __launch_bounds__(256) void mpo_pair_merge(
    const float* __restrict__ c0, const float* __restrict__ c1,
    const float* __restrict__ c2, const float* __restrict__ c3,
    float* __restrict__ W01, float* __restrict__ W23) {
  int idx = blockIdx.x * 256 + threadIdx.x;        // 0 .. 2*262144
  int t    = idx & 0x3FFFF;
  int lo6  = t & 63;
  int mid6 = (t >> 6) & 63;
  int hi6  = t >> 12;
  float acc = 0.f;
  if (idx < 262144) {                              // W01[A,C,r2]
    int r2 = lo6, C = mid6, A = hi6;
    int o1 = A >> 3, o2 = A & 7, i1 = C >> 3, i2 = C & 7;
    const float* p0 = c0 + (o1 * 8 + i1) * 64;                 // + r1
    const float* p1 = c1 + (o2 * 8 + i2) * 64 + r2;            // + r1*4096
#pragma unroll 8
    for (int r1 = 0; r1 < 64; ++r1) acc += p0[r1] * p1[r1 * 4096];
    W01[(A * 64 + C) * 64 + r2] = acc;
  } else {                                         // W23[r2,B,D]
    int D = lo6, B = mid6, r2 = hi6;
    int o3 = B >> 3, o4 = B & 7, i3 = D >> 3, i4 = D & 7;
    const float* p2 = c2 + ((r2 * 8 + o3) * 8 + i3) * 64;      // + r3
    const float* p3 = c3 + (o4 * 8 + i4);                      // + r3*64
#pragma unroll 8
    for (int r3 = 0; r3 < 64; ++r3) acc += p2[r3] * p3[r3 * 64];
    W23[(r2 * 64 + B) * 64 + D] = acc;
  }
}

// ------------------- Phase 2: dense W, bf16 hi/lo split ---------------------
// W[O,I] = sum_r2 W01[A,C,r2] * W23[r2,B,D];  O=A*64+B, I=C*64+D (row-major)
__global__ __launch_bounds__(256) void build_w_split(
    const float* __restrict__ W01, const float* __restrict__ W23,
    bf16_t* __restrict__ Whi, bf16_t* __restrict__ Wlo) {
  int idx = blockIdx.x * 256 + threadIdx.x;        // 0 .. 16777216
  int I = idx & 4095, O = idx >> 12;
  int A = O >> 6, B = O & 63, C = I >> 6, D = I & 63;
  const float* pa = W01 + (A * 64 + C) * 64;       // contiguous in r2
  const float* pb = W23 + B * 64 + D;              // + r2*4096
  float acc = 0.f;
#pragma unroll 16
  for (int r2 = 0; r2 < 64; ++r2) acc += pa[r2] * pb[r2 * 4096];
  bf16_t hi = f32_to_bf16_rne(acc);
  Whi[idx] = hi;                                   // re-read by GEMM: keep RT
  Wlo[idx] = f32_to_bf16_rne(acc - bf16_to_f32(hi));
}

// --------------------------- Phase 3: WMMA GEMM -----------------------------
// out[t,O] = sum_I x[t,I] * W[O,I] + bias[O]
// Block: 128 thr (4 waves, 2x2), tile 128(M) x 128(N); wave tile 64x64.
// K-step 32: 48 v_wmma + 32 b128 loads, uniform-base + imm-offset addressing.
__global__ __launch_bounds__(128) void mpo_gemm_wmma(
    const bf16_t* __restrict__ Xhi, const bf16_t* __restrict__ Xlo,
    const bf16_t* __restrict__ Whi, const bf16_t* __restrict__ Wlo,
    const float* __restrict__ bias, float* __restrict__ out) {
  const int K = 4096;
  const int lane = threadIdx.x & 31;
  const int wave = threadIdx.x >> 5;               // 0..3
  const int half = lane >> 4;                      // lane group 0/1
  const int lm   = lane & 15;
  const int wm = wave & 1;                         // 2 waves along M
  const int wn = wave >> 1;                        // 2 waves along N
  const int mBase = blockIdx.y * 128 + wm * 64;
  const int nBase = blockIdx.x * 128 + wn * 64;

  // Per-lane 32-bit element offsets (uniform base + voffset + imm addressing).
  // A 16x32 layout: lane holds row lm; K = kb + half*8 + {0..7},
  //                                    then kb + 16 + half*8 + {0..7}
  int aoff[4];
#pragma unroll
  for (int mi = 0; mi < 4; ++mi)
    aoff[mi] = (mBase + mi * 16 + lm) * K + half * 8;
  // B 32x16 layout: lane holds col lm; K = kb + half*8 + {0..7}, then kb + 16 + half*8 + {0..7}
  int boff[4];
#pragma unroll
  for (int ni = 0; ni < 4; ++ni)
    boff[ni] = (nBase + ni * 16 + lm) * K + half * 8;

  v8f acc[4][4] = {};

  for (int kb = 0; kb < K; kb += 32) {
    // B fragments live across the whole k-step (reused by 4 M-subtiles)
    Frag16 bhi[4], blo[4];
#pragma unroll
    for (int ni = 0; ni < 4; ++ni) {
      bhi[ni].h[0] = *(const v8bf*)(Whi + (boff[ni] + kb));
      bhi[ni].h[1] = *(const v8bf*)(Whi + (boff[ni] + kb + 16));
      blo[ni].h[0] = *(const v8bf*)(Wlo + (boff[ni] + kb));
      blo[ni].h[1] = *(const v8bf*)(Wlo + (boff[ni] + kb + 16));
    }
    // A fragments loaded per mi (short live range), reused by 4 N-subtiles
#pragma unroll
    for (int mi = 0; mi < 4; ++mi) {
      Frag16 ahi, alo;
      ahi.h[0] = *(const v8bf*)(Xhi + (aoff[mi] + kb));
      ahi.h[1] = *(const v8bf*)(Xhi + (aoff[mi] + kb + 16));
      alo.h[0] = *(const v8bf*)(Xlo + (aoff[mi] + kb));
      alo.h[1] = *(const v8bf*)(Xlo + (aoff[mi] + kb + 16));
      // 3-pass split-precision MACs, fp32 accumulate (lo*lo negligible)
#pragma unroll
      for (int ni = 0; ni < 4; ++ni) {
        acc[mi][ni] = __builtin_amdgcn_wmma_f32_16x16x32_bf16(
            false, ahi.v, false, bhi[ni].v, (short)0, acc[mi][ni], false, false);
        acc[mi][ni] = __builtin_amdgcn_wmma_f32_16x16x32_bf16(
            false, ahi.v, false, blo[ni].v, (short)0, acc[mi][ni], false, false);
        acc[mi][ni] = __builtin_amdgcn_wmma_f32_16x16x32_bf16(
            false, alo.v, false, bhi[ni].v, (short)0, acc[mi][ni], false, false);
      }
    }
  }

  // epilogue: D layout VGPR v -> row v + 8*half, col lm; NT stores (no reuse)
#pragma unroll
  for (int mi = 0; mi < 4; ++mi)
#pragma unroll
    for (int ni = 0; ni < 4; ++ni) {
      int col = nBase + ni * 16 + lm;
      float b = bias[col];
#pragma unroll
      for (int v = 0; v < 8; ++v) {
        int row = mBase + mi * 16 + v + half * 8;
        __builtin_nontemporal_store(acc[mi][ni][v] + b,
                                    out + (size_t)row * 4096 + col);
      }
    }
}

// ---------------------------------------------------------------------------
extern "C" void kernel_launch(void* const* d_in, const int* in_sizes, int n_in,
                              void* d_out, int out_size, void* d_ws, size_t ws_size,
                              hipStream_t stream) {
  const float* x    = (const float*)d_in[0];
  const float* c0   = (const float*)d_in[1];
  const float* c1   = (const float*)d_in[2];
  const float* c2   = (const float*)d_in[3];
  const float* c3   = (const float*)d_in[4];
  const float* bias = (const float*)d_in[5];
  float* out = (float*)d_out;

  // ws layout: W01(1MB) | W23(1MB) | Whi(32MB) | Wlo(32MB) | Xhi(32MB) | Xlo(32MB)
  const size_t NELEM = (size_t)4096 * 4096;
  float*  W01 = (float*)d_ws;
  float*  W23 = W01 + 64 * 64 * 64;
  bf16_t* Whi = (bf16_t*)(W23 + 64 * 64 * 64);
  bf16_t* Wlo = Whi + NELEM;
  bf16_t* Xhi = Wlo + NELEM;
  bf16_t* Xlo = Xhi + NELEM;

  split_x<<<16384, 256, 0, stream>>>(x, Xhi, Xlo);
  mpo_pair_merge<<<2048, 256, 0, stream>>>(c0, c1, c2, c3, W01, W23);
  build_w_split<<<65536, 256, 0, stream>>>(W01, W23, Whi, Wlo);
  dim3 grid(32, 32);  // N-tiles x M-tiles
  mpo_gemm_wmma<<<grid, 128, 0, stream>>>(Xhi, Xlo, Whi, Wlo, bias, out);
}
